// Head_31748398252101
// MI455X (gfx1250) — matrix operand order repaired and checked by reading, not compiled, not run on hardware
//
#include <hip/hip_runtime.h>

typedef __attribute__((ext_vector_type(16))) _Float16 v16h;
typedef __attribute__((ext_vector_type(8)))  _Float16 v8h;
typedef __attribute__((ext_vector_type(8)))  float    v8f;
typedef __attribute__((ext_vector_type(4)))  float    v4f;

#define NB   4
#define SEQ  4096
#define EMB  1024
#define HD   64

__device__ __forceinline__ v8f wmma_f16(v16h a, v16h b, v8f c) {
  return __builtin_amdgcn_wmma_f32_16x16x32_f16(false, a, false, b, (short)0, c, false, false);
}

// ---------------------------------------------------------------------------
// Kernel 0: convert Wq, Wk (f32 [64][1024]) to f16 staging buffers.
// 8 elements per thread, disjoint ranges, vector stores.
// ---------------------------------------------------------------------------
__global__ __launch_bounds__(256) void cvt_w(
    const float* __restrict__ Wq, const float* __restrict__ Wk,
    _Float16* __restrict__ Wqh, _Float16* __restrict__ Wkh)
{
  const int i = (blockIdx.x * 256 + threadIdx.x) * 8;   // 65536 elems / matrix
  v4f q0 = *(const v4f*)(Wq + i), q1 = *(const v4f*)(Wq + i + 4);
  v4f k0 = *(const v4f*)(Wk + i), k1 = *(const v4f*)(Wk + i + 4);
  v8h oq, ok;
#pragma unroll
  for (int j = 0; j < 4; ++j) {
    oq[j] = (_Float16)q0[j]; oq[j + 4] = (_Float16)q1[j];
    ok[j] = (_Float16)k0[j]; ok[j + 4] = (_Float16)k1[j];
  }
  *(v8h*)(Wqh + i) = oq;
  *(v8h*)(Wkh + i) = ok;
}

// ---------------------------------------------------------------------------
// Kernel 1: Q = x*Wq^T, K = x*Wk^T  (V == K per reference).
// One wave computes a 16x64 tile of both Q and K. Software-pipelined k-loop.
// Outputs: Qh[s][64], Kh[s][64], Kth[b][64][4096]  (all f16).
// ---------------------------------------------------------------------------
__global__ __launch_bounds__(128) void proj_qk(
    const float* __restrict__ x, const _Float16* __restrict__ Wqh,
    const _Float16* __restrict__ Wkh, _Float16* __restrict__ Qh,
    _Float16* __restrict__ Kh, _Float16* __restrict__ Kth)
{
  const int lane = threadIdx.x & 31;
  const int wave = threadIdx.x >> 5;
  const int mt   = blockIdx.x * 4 + wave;   // 0..1023 (16384/16 row tiles)
  const int m0   = mt * 16;
  const int lr   = lane & 15;
  const int lh   = lane >> 4;

  const int arow = m0 + lr;
  const float*    xbase = x + (size_t)arow * EMB + lh * 16;
  const _Float16* wqb   = Wqh + (size_t)lr * EMB + lh * 16;
  const _Float16* wkb   = Wkh + (size_t)lr * EMB + lh * 16;

  v8f accQ[4] = {}, accK[4] = {};

  // fragment loaders --------------------------------------------------------
  auto loadB = [&](int kt, v16h* bq, v16h* bk) {
#pragma unroll
    for (int nt = 0; nt < 4; ++nt) {
      bq[nt] = *(const v16h*)(wqb + (size_t)nt * 16 * EMB + kt);
      bk[nt] = *(const v16h*)(wkb + (size_t)nt * 16 * EMB + kt);
    }
  };
  auto loadA = [&](int kt) -> v16h {
    const float* xr = xbase + kt;
    v4f f0 = *(const v4f*)(xr);
    v4f f1 = *(const v4f*)(xr + 4);
    v4f f2 = *(const v4f*)(xr + 8);
    v4f f3 = *(const v4f*)(xr + 12);
    v16h a;
#pragma unroll
    for (int j = 0; j < 4; ++j) {
      a[j]      = (_Float16)f0[j];
      a[j + 4]  = (_Float16)f1[j];
      a[j + 8]  = (_Float16)f2[j];
      a[j + 12] = (_Float16)f3[j];
    }
    return a;
  };

  // software pipeline: prefetch next k-step before computing current --------
  v16h bq[4], bk[4], a;
  loadB(0, bq, bk);
  a = loadA(0);

#pragma unroll 2
  for (int kt = 0; kt < EMB; kt += 32) {
    const int ktn = (kt + 32) & (EMB - 1);   // wraps on last iter (harmless prefetch)
    v16h nbq[4], nbk[4];
    loadB(ktn, nbq, nbk);
    v16h na = loadA(ktn);

#pragma unroll
    for (int nt = 0; nt < 4; ++nt) {
      accQ[nt] = wmma_f16(a, bq[nt], accQ[nt]);
      accK[nt] = wmma_f16(a, bk[nt], accK[nt]);
    }
#pragma unroll
    for (int nt = 0; nt < 4; ++nt) { bq[nt] = nbq[nt]; bk[nt] = nbk[nt]; }
    a = na;
  }

  // C/D layout: VGPR r, lanes<16 -> M=r, lanes>=16 -> M=r+8; N = lane%16
#pragma unroll
  for (int nt = 0; nt < 4; ++nt) {
#pragma unroll
    for (int r = 0; r < 8; ++r) {
      const int row = m0 + r + lh * 8;
      const int col = nt * 16 + lr;
      const _Float16 q = (_Float16)accQ[nt][r];
      const _Float16 k = (_Float16)accK[nt][r];
      Qh[(size_t)row * HD + col] = q;
      Kh[(size_t)row * HD + col] = k;
      const int b = row >> 12, s = row & (SEQ - 1);
      Kth[((size_t)b * HD + col) * SEQ + s] = k;
    }
  }
}

// ---------------------------------------------------------------------------
// Kernel 2: causal flash attention, one wave per 16-query tile, 32-key steps.
// ---------------------------------------------------------------------------
__global__ __launch_bounds__(128) void flash_attn(
    const _Float16* __restrict__ Qh, const _Float16* __restrict__ Kh,
    const _Float16* __restrict__ Kth, float* __restrict__ out)
{
  __shared__ __align__(16) _Float16 pbuf[4][16][40]; // padded stride: no bank conflicts

  const int lane = threadIdx.x & 31;
  const int wave = threadIdx.x >> 5;
  const int qt   = blockIdx.x * 4 + wave;   // 0..1023
  const int b    = qt >> 8;
  const int t    = qt & 255;
  const int q0   = t * 16;
  const int lr   = lane & 15;
  const int lh   = lane >> 4;

  // Q as A operand (d contraction split into two 32-chunks)
  v16h qa[2];
  {
    const _Float16* qrow = Qh + ((size_t)b * SEQ + q0 + lr) * HD;
#pragma unroll
    for (int c = 0; c < 2; ++c) {
      const int db = c * 32 + lh * 16;
      qa[c] = *(const v16h*)(qrow + db);
    }
  }

  // all-ones B operand: row-sum accumulator via WMMA (softmax denominator)
  v16h vones;
#pragma unroll
  for (int j = 0; j < 16; ++j) vones[j] = (_Float16)1.0f;

  const _Float16* kbase = Kh  + ((size_t)b * SEQ + lr) * HD;
  const _Float16* vbase = Kth + ((size_t)b * HD + lr) * SEQ + lh * 16;

  v8f o[4] = {};
  v8f lacc = {};
  float mrow[8];
#pragma unroll
  for (int r = 0; r < 8; ++r) mrow[r] = -1e30f;

  const int nk     = q0 + 16;               // valid keys (causal)
  const int nsteps = (nk + 31) >> 5;

  for (int stp = 0; stp < nsteps; ++stp) {
    const int kb = stp * 32;

    // ---- issue ALL global loads for this step up front ----
    // K fragments for S = Q K^T (two 16-key sub-tiles x two d-chunks)
    v16h kf[4];
    {
      const _Float16* k0p = kbase + (size_t)kb * HD;
      const _Float16* k1p = k0p + (size_t)16 * HD;   // kb+31 <= 4095 always
#pragma unroll
      for (int c = 0; c < 2; ++c) {
        const int db = c * 32 + lh * 16;
        kf[c]     = *(const v16h*)(k0p + db);
        kf[c + 2] = *(const v16h*)(k1p + db);
      }
    }
    // V fragments for PV (latency hidden behind the softmax phase below)
    v16h vb[4];
#pragma unroll
    for (int nt = 0; nt < 4; ++nt)
      vb[nt] = *(const v16h*)(vbase + (size_t)nt * 16 * SEQ + kb);

    // ---- S WMMAs ----
    v8f s0 = {}, s1 = {};
#pragma unroll
    for (int c = 0; c < 2; ++c) {
      s0 = wmma_f16(qa[c], kf[c],     s0);
      s1 = wmma_f16(qa[c], kf[c + 2], s1);
    }

    // ---- online softmax (per row r owned by this half's 16 lanes) ----
#pragma unroll
    for (int r = 0; r < 8; ++r) {
      const int q = q0 + r + lh * 8;
      float a0 = s0[r] * 0.125f;             // 1/sqrt(64)
      float a1 = s1[r] * 0.125f;
      if (kb + lr      > q) a0 = -1e30f;     // causal mask (also kills tail keys)
      if (kb + 16 + lr > q) a1 = -1e30f;

      float mx = fmaxf(a0, a1);
#pragma unroll
      for (int off = 1; off < 16; off <<= 1)
        mx = fmaxf(mx, __shfl_xor(mx, off, 32));
      const float mn = fmaxf(mrow[r], mx);
      const float cr = __expf(mrow[r] - mn);
      const float e0 = __expf(a0 - mn);
      const float e1 = __expf(a1 - mn);

      o[0][r] *= cr; o[1][r] *= cr; o[2][r] *= cr; o[3][r] *= cr;
      lacc[r] *= cr;
      mrow[r] = mn;

      // P to LDS (row = query-in-tile, col = key-in-step)
      pbuf[wave][r + lh * 8][lr]      = (_Float16)e0;
      pbuf[wave][r + lh * 8][16 + lr] = (_Float16)e1;
    }

    // same-wave DS store->load ordering barrier
    asm volatile("s_wait_dscnt 0" ::: "memory");

    // ---- read P back in A-operand layout ----
    v16h pa;
    {
      const _Float16* prow = &pbuf[wave][lr][lh * 16];
      v8h lo = *(const v8h*)(prow);        // keys lh*16 + 0..7
      v8h hi = *(const v8h*)(prow + 8);    // keys lh*16 + 8..15
#pragma unroll
      for (int j = 0; j < 8; ++j) { pa[j] = lo[j]; pa[j + 8] = hi[j]; }
    }

    // ---- O += P*V (V == K), denominator += P*ones ----
#pragma unroll
    for (int nt = 0; nt < 4; ++nt)
      o[nt] = wmma_f16(pa, vb[nt], o[nt]);
    lacc = wmma_f16(pa, vones, lacc);      // row sums (replicated across lanes)
  }

  // ---- epilogue: normalize and store f32 ----
#pragma unroll
  for (int nt = 0; nt < 4; ++nt) {
#pragma unroll
    for (int r = 0; r < 8; ++r) {
      const int row = b * SEQ + q0 + r + lh * 8;
      out[(size_t)row * HD + nt * 16 + lr] = o[nt][r] / lacc[r];
    }
  }
}

// ---------------------------------------------------------------------------
extern "C" void kernel_launch(void* const* d_in, const int* in_sizes, int n_in,
                              void* d_out, int out_size, void* d_ws, size_t ws_size,
                              hipStream_t stream) {
  (void)in_sizes; (void)n_in; (void)out_size; (void)ws_size;
  const float* x  = (const float*)d_in[0];
  const float* Wq = (const float*)d_in[1];
  const float* Wk = (const float*)d_in[2];
  // d_in[3] (Wv) intentionally unused: reference computes V with Wk.

  _Float16* Qh  = (_Float16*)d_ws;                       // 2 MB
  _Float16* Kh  = Qh + (size_t)NB * SEQ * HD;            // 2 MB
  _Float16* Kth = Kh + (size_t)NB * SEQ * HD;            // 2 MB
  _Float16* Wqh = Kth + (size_t)NB * SEQ * HD;           // 128 KB
  _Float16* Wkh = Wqh + (size_t)HD * EMB;                // 128 KB
  float* out = (float*)d_out;

  cvt_w<<<32, 256, 0, stream>>>(Wq, Wk, Wqh, Wkh);       // 65536/(256*8) = 32 blocks
  proj_qk<<<256, 128, 0, stream>>>(x, Wqh, Wkh, Qh, Kh, Kth);
  flash_attn<<<256, 128, 0, stream>>>(Qh, Kh, Kth, out);
}
